// GINEEncoder_77541339562639
// MI455X (gfx1250) — matrix operand order repaired and checked by reading, not compiled, not run on hardware
//
#include <hip/hip_runtime.h>

// ---------------------------------------------------------------------------
// GINE encoder for MI455X (gfx1250, wave32).
// Bandwidth-bound: e stream stored bf16, GEMMs via v_wmma_f32_16x16x32_bf16,
// h/agg/pooled kept f32 (L2-resident), W1||W2 staged to LDS (TDM + vector).
// ---------------------------------------------------------------------------

typedef unsigned int u32;
typedef unsigned short u16;
typedef __bf16 bf16;
typedef __attribute__((ext_vector_type(16))) __bf16 v16bf;
typedef __attribute__((ext_vector_type(4)))  __bf16 v4bf;
typedef __attribute__((ext_vector_type(8)))  float  v8f;
typedef __attribute__((ext_vector_type(4)))  u32    su32x4;
typedef __attribute__((ext_vector_type(8)))  u32    su32x8;

#define N_NODES  40000
#define N_EDGES  640000
#define N_GRAPHS 64
#define T_STEPS  4

union ABF { bf16 e[16]; v16bf v; uint4 q[2]; };

__device__ __forceinline__ u16 f2bf_bits(float f) {
  union { float f; u32 u; } c; c.f = f;
  u32 u = c.u;
  u32 r = u + 0x7FFFu + ((u >> 16) & 1u);   // round-to-nearest-even
  return (u16)(r >> 16);
}

__device__ __forceinline__ v8f wmma_bf16(v16bf a, v16bf b, v8f c) {
  return __builtin_amdgcn_wmma_f32_16x16x32_bf16(false, a, false, b, (short)0, c,
                                                 false, false);
}

// --- Tensor Data Mover: 1D dword copy global -> LDS (inline asm, portable) --
// lds_dst's address feeds the descriptor AND an asm operand, so the LDS
// object visibly escapes (prevents "no stores -> loads are undef" folding).
__device__ __forceinline__ void tdm_load_to_lds_1d(const void* src, void* lds_dst,
                                                   u32 ndwords) {
  unsigned long long ga = (unsigned long long)(size_t)src;
  u32 lds_byte = (u32)(size_t)lds_dst;      // low 32 bits of LDS generic addr
  su32x4 g0;
  g0[0] = 1u;                                            // count=1 valid
  g0[1] = lds_byte;                                      // lds_addr
  g0[2] = (u32)ga;                                       // global_addr[31:0]
  g0[3] = ((u32)(ga >> 32) & 0x01FFFFFFu) | 0x80000000u; // addr[56:32] | type=2
  su32x8 g1;
  g1[0] = 2u << 16;                                      // data_size = 4B
  g1[1] = (ndwords & 0xFFFFu) << 16;                     // tensor_dim0 lo16
  g1[2] = (ndwords >> 16) & 0xFFFFu;                     // tensor_dim0 hi16
  g1[3] = (ndwords & 0xFFFFu) << 16;                     // tile_dim0
  g1[4] = 0u;                                            // tile_dim1/2 unused
  g1[5] = ndwords;                                       // tensor_dim0_stride lo
  g1[6] = 0u;
  g1[7] = 0u;
  asm volatile("tensor_load_to_lds %0, %1"
               :: "s"(g0), "s"(g1), "v"(lds_dst)
               : "memory");
}

// --- A-operand builders (16-bit A 16x32 lane layout, §7.12.2) ---------------
__device__ __forceinline__ v16bf a_from_f32(const float* __restrict__ p, int c0,
                                            int c1) {
  float4 p0 = *(const float4*)(p + c0);
  float4 p1 = *(const float4*)(p + c0 + 4);
  float4 p2 = *(const float4*)(p + c1);
  float4 p3 = *(const float4*)(p + c1 + 4);
  ABF a;
  a.e[0]  = (bf16)p0.x; a.e[1]  = (bf16)p0.y; a.e[2]  = (bf16)p0.z; a.e[3]  = (bf16)p0.w;
  a.e[4]  = (bf16)p1.x; a.e[5]  = (bf16)p1.y; a.e[6]  = (bf16)p1.z; a.e[7]  = (bf16)p1.w;
  a.e[8]  = (bf16)p2.x; a.e[9]  = (bf16)p2.y; a.e[10] = (bf16)p2.z; a.e[11] = (bf16)p2.w;
  a.e[12] = (bf16)p3.x; a.e[13] = (bf16)p3.y; a.e[14] = (bf16)p3.z; a.e[15] = (bf16)p3.w;
  return a.v;
}

// z = h+agg with fused agg reset (vectorized 16B traffic)
__device__ __forceinline__ v16bf a_from_hz(float* __restrict__ hp,
                                           float* __restrict__ ap, int c0, int c1) {
  float4 h0 = *(const float4*)(hp + c0);
  float4 h1 = *(const float4*)(hp + c0 + 4);
  float4 h2 = *(const float4*)(hp + c1);
  float4 h3 = *(const float4*)(hp + c1 + 4);
  float4 g0 = *(const float4*)(ap + c0);
  float4 g1 = *(const float4*)(ap + c0 + 4);
  float4 g2 = *(const float4*)(ap + c1);
  float4 g3 = *(const float4*)(ap + c1 + 4);
  const float4 z4 = {0.f, 0.f, 0.f, 0.f};
  *(float4*)(ap + c0)     = z4;
  *(float4*)(ap + c0 + 4) = z4;
  *(float4*)(ap + c1)     = z4;
  *(float4*)(ap + c1 + 4) = z4;
  ABF a;
  a.e[0]  = (bf16)(h0.x + g0.x); a.e[1]  = (bf16)(h0.y + g0.y);
  a.e[2]  = (bf16)(h0.z + g0.z); a.e[3]  = (bf16)(h0.w + g0.w);
  a.e[4]  = (bf16)(h1.x + g1.x); a.e[5]  = (bf16)(h1.y + g1.y);
  a.e[6]  = (bf16)(h1.z + g1.z); a.e[7]  = (bf16)(h1.w + g1.w);
  a.e[8]  = (bf16)(h2.x + g2.x); a.e[9]  = (bf16)(h2.y + g2.y);
  a.e[10] = (bf16)(h2.z + g2.z); a.e[11] = (bf16)(h2.w + g2.w);
  a.e[12] = (bf16)(h3.x + g3.x); a.e[13] = (bf16)(h3.y + g3.y);
  a.e[14] = (bf16)(h3.z + g3.z); a.e[15] = (bf16)(h3.w + g3.w);
  return a.v;
}

// one K-step over all 8 N-tiles (called with literal kt -> guaranteed unroll)
__device__ __forceinline__ void gemm_step(v8f (&acc)[8], v16bf a,
                                          const bf16* __restrict__ Wlds,
                                          int kt, int lane) {
#pragma unroll
  for (int nt = 0; nt < 8; ++nt) {
    v16bf b = *(const v16bf*)(Wlds + (size_t)((nt * 4 + kt) * 32 + lane) * 16);
    acc[nt] = wmma_bf16(a, b, acc[nt]);
  }
}

// --- h = relu((h+agg)@W1+b1)@W2+b2 ; agg=0 ; pooled[g, t*128+:] += h --------
__global__ __launch_bounds__(256) void mlp_kernel(
    const bf16* __restrict__ Wp, const float* __restrict__ b1,
    const float* __restrict__ b2, float* __restrict__ h,
    float* __restrict__ agg, const int* __restrict__ gids,
    float* __restrict__ pooled, int t, int nTiles) {
  __shared__ __align__(32) bf16 ldsW[32768];      // W1p (16384) || W2p (16384)
  __shared__ __align__(32) bf16 ldsY[8][2048];    // per-wave 16x128 bf16 bounce

  // Stage 64KB weights: TDM (authoritative DMA path) + cooperative vector
  // copy of the same bytes (keeps compiler's view of LDS contents defined).
  {
    const uint4* gsrc = (const uint4*)Wp;
    uint4* ldst = (uint4*)ldsW;
#pragma unroll
    for (int i = 0; i < 16; ++i) {
      int idx = threadIdx.x + i * 256;            // 4096 x 16B = 64KB
      ldst[idx] = gsrc[idx];
    }
  }
  if (threadIdx.x == 0) {
    tdm_load_to_lds_1d(Wp, ldsW, 16384u);
    __builtin_amdgcn_s_wait_tensorcnt(0);
  }
  __syncthreads();

  int wave = threadIdx.x >> 5, lane = threadIdx.x & 31;
  int tile = blockIdx.x * 8 + wave;
  if (tile >= nTiles) return;
  int half = lane >> 4, rsel = lane & 15;
  int rowA = tile * 16 + rsel;
  float* hp = h   + (size_t)rowA * 128;
  float* ap = agg + (size_t)rowA * 128;

  // GEMM1: explicit 4x k-steps, each 8 WMMAs (fully flat schedule)
  v8f acc[8] = {};
  {
    v16bf a;
    a = a_from_hz(hp, ap, 0 * 32 + half * 8, 0 * 32 + 16 + half * 8);
    gemm_step(acc, a, ldsW, 0, lane);
    a = a_from_hz(hp, ap, 1 * 32 + half * 8, 1 * 32 + 16 + half * 8);
    gemm_step(acc, a, ldsW, 1, lane);
    a = a_from_hz(hp, ap, 2 * 32 + half * 8, 2 * 32 + 16 + half * 8);
    gemm_step(acc, a, ldsW, 2, lane);
    a = a_from_hz(hp, ap, 3 * 32 + half * 8, 3 * 32 + 16 + half * 8);
    gemm_step(acc, a, ldsW, 3, lane);
  }
  // bias + relu -> bf16 bounce tile (D layout -> row-major in LDS)
#pragma unroll
  for (int nt = 0; nt < 8; ++nt) {
    int n = nt * 16 + rsel;
    float bias = b1[n];
#pragma unroll
    for (int r = 0; r < 8; ++r) {
      int M = r + half * 8;
      ldsY[wave][M * 128 + n] = (bf16)fmaxf(acc[nt][r] + bias, 0.f);
    }
  }
  // GEMM2: A re-read from LDS as two 16B ds loads per k-step
  v8f acc2[8] = {};
  const bf16* yp = &ldsY[wave][rsel * 128];
  {
    ABF a;
    a.q[0] = *(const uint4*)(yp + 0 * 32 + half * 8);
    a.q[1] = *(const uint4*)(yp + 0 * 32 + 16 + half * 8);
    gemm_step(acc2, a.v, ldsW + 16384, 0, lane);
    a.q[0] = *(const uint4*)(yp + 1 * 32 + half * 8);
    a.q[1] = *(const uint4*)(yp + 1 * 32 + 16 + half * 8);
    gemm_step(acc2, a.v, ldsW + 16384, 1, lane);
    a.q[0] = *(const uint4*)(yp + 2 * 32 + half * 8);
    a.q[1] = *(const uint4*)(yp + 2 * 32 + 16 + half * 8);
    gemm_step(acc2, a.v, ldsW + 16384, 2, lane);
    a.q[0] = *(const uint4*)(yp + 3 * 32 + half * 8);
    a.q[1] = *(const uint4*)(yp + 3 * 32 + 16 + half * 8);
    gemm_step(acc2, a.v, ldsW + 16384, 3, lane);
  }
  // epilogue: new h (f32), graph pooling (f32 atomics, 128KB L2-resident)
  float bias2[8];
#pragma unroll
  for (int nt = 0; nt < 8; ++nt) bias2[nt] = b2[nt * 16 + rsel];
#pragma unroll
  for (int r = 0; r < 8; ++r) {
    int M = r + half * 8;
    int node = tile * 16 + M;
    int g = gids[node];
    float* hrow = h + (size_t)node * 128;
    float* prow = pooled + (size_t)g * 512 + t * 128;
#pragma unroll
    for (int nt = 0; nt < 8; ++nt) {
      int n = nt * 16 + rsel;
      float v = acc2[nt][r] + bias2[nt];
      hrow[n] = v;
      atomicAdd(prow + n, v);
    }
  }
}

// --- h = x @ Wn + bn  (x:[N,32] f32 -> h:[N,128] f32) -----------------------
__global__ __launch_bounds__(256) void node_transform_kernel(
    const float* __restrict__ x, const bf16* __restrict__ Wnp,
    const float* __restrict__ bn, float* __restrict__ h, int nTiles) {
  int wave = threadIdx.x >> 5, lane = threadIdx.x & 31;
  int tile = blockIdx.x * 8 + wave;
  if (tile >= nTiles) return;
  int half = lane >> 4, rsel = lane & 15;
  int row = tile * 16 + rsel;
  v16bf a = a_from_f32(x + (size_t)row * 32, half * 8, 16 + half * 8);
  v8f acc[8] = {};
#pragma unroll
  for (int nt = 0; nt < 8; ++nt) {
    v16bf b = *(const v16bf*)(Wnp + (size_t)(nt * 32 + lane) * 16);
    acc[nt] = wmma_bf16(a, b, acc[nt]);
  }
#pragma unroll
  for (int nt = 0; nt < 8; ++nt) {
    int n = nt * 16 + rsel;
    float bias = bn[n];
#pragma unroll
    for (int r = 0; r < 8; ++r) {
      int M = r + half * 8;
      h[(size_t)(tile * 16 + M) * 128 + n] = acc[nt][r] + bias;
    }
  }
}

// --- e = edge_attr @ We + be  (f32[E,16] -> bf16[E,128]) --------------------
__global__ __launch_bounds__(256) void edge_transform_kernel(
    const float* __restrict__ ea, const bf16* __restrict__ Wep,
    const float* __restrict__ be, bf16* __restrict__ e, int nTiles) {
  int wave = threadIdx.x >> 5, lane = threadIdx.x & 31;
  int tile = blockIdx.x * 8 + wave;
  if (tile >= nTiles) return;
  int half = lane >> 4, rsel = lane & 15;
  int row = tile * 16 + rsel;
  const float* p = ea + (size_t)row * 16 + half * 8;
  float4 p0 = *(const float4*)(p);
  float4 p1 = *(const float4*)(p + 4);
  ABF a;                                          // K padded 16 -> 32 (zeros)
  a.e[0] = (bf16)p0.x; a.e[1] = (bf16)p0.y; a.e[2] = (bf16)p0.z; a.e[3] = (bf16)p0.w;
  a.e[4] = (bf16)p1.x; a.e[5] = (bf16)p1.y; a.e[6] = (bf16)p1.z; a.e[7] = (bf16)p1.w;
#pragma unroll
  for (int j = 8; j < 16; ++j) a.e[j] = (bf16)0.f;
  v8f acc[8] = {};
#pragma unroll
  for (int nt = 0; nt < 8; ++nt) {
    v16bf b = *(const v16bf*)(Wep + (size_t)(nt * 32 + lane) * 16);
    acc[nt] = wmma_bf16(a.v, b, acc[nt]);
  }
#pragma unroll
  for (int nt = 0; nt < 8; ++nt) {
    int n = nt * 16 + rsel;
    float bias = be[n];
#pragma unroll
    for (int r = 0; r < 8; ++r) {
      int M = r + half * 8;
      e[(size_t)(tile * 16 + M) * 128 + n] = (bf16)(acc[nt][r] + bias);
    }
  }
}

// --- msg = relu(h[src]+e); agg[dst] += msg (f32 atomics, L2-resident) -------
__global__ __launch_bounds__(256) void edge_msg_kernel(
    const bf16* __restrict__ e, const float* __restrict__ h,
    const int* __restrict__ ei, float* __restrict__ agg, int E) {
  int gid = blockIdx.x * 256 + threadIdx.x;
  int edge = gid >> 5, lane = gid & 31;            // one wave per edge, 4 ch/lane
  if (edge >= E) return;
  int src = ei[edge];
  int dst = ei[E + edge];
  v4bf ev = *(const v4bf*)(e + (size_t)edge * 128 + lane * 4);
  const float4 hv = *(const float4*)(h + (size_t)src * 128 + lane * 4);
  float m0 = fmaxf(hv.x + (float)ev[0], 0.f);
  float m1 = fmaxf(hv.y + (float)ev[1], 0.f);
  float m2 = fmaxf(hv.z + (float)ev[2], 0.f);
  float m3 = fmaxf(hv.w + (float)ev[3], 0.f);
  float* ag = agg + (size_t)dst * 128 + lane * 4;
  atomicAdd(ag + 0, m0); atomicAdd(ag + 1, m1);
  atomicAdd(ag + 2, m2); atomicAdd(ag + 3, m3);
}

// --- pack a [K,128] f32 weight into WMMA-B lane layout (bf16 tiles) ---------
// tile index = nt*kTiles + kt ; within tile: lane 0..31 x 8 dwords.
// lane<16: column n=nt*16+lane, K = kt*32 + 0..7,16..23 ; lane>=16: K = kt*32+8..15,24..31
__global__ __launch_bounds__(256) void pack_w_kernel(const float* __restrict__ W,
                                                     u32* __restrict__ dst,
                                                     int K, int kTiles, int total) {
  int id = blockIdx.x * 256 + threadIdx.x;
  if (id >= total) return;
  int v = id & 7;
  int lane = (id >> 3) & 31;
  int tileIdx = id >> 8;
  int nt = tileIdx / kTiles, kt = tileIdx % kTiles;
  int half = lane >> 4;
  int n = nt * 16 + (lane & 15);
  int k0 = kt * 32 + (v >> 2) * 16 + half * 8 + 2 * (v & 3);
  float lo = (k0     < K) ? W[(size_t)k0 * 128 + n]       : 0.f;
  float hi = (k0 + 1 < K) ? W[(size_t)(k0 + 1) * 128 + n] : 0.f;
  dst[id] = (u32)f2bf_bits(lo) | ((u32)f2bf_bits(hi) << 16);
}

__global__ __launch_bounds__(256) void zero_kernel(float* __restrict__ p, long n) {
  long i = (long)blockIdx.x * 256 + threadIdx.x;
  if (i < n) p[i] = 0.f;
}

// --- out = pooled @ Wl + bl  ([64,512]@[512,256], tiny, full f32) -----------
__global__ __launch_bounds__(256) void final_kernel(
    const float* __restrict__ pooled, const float* __restrict__ Wl,
    const float* __restrict__ bl, float* __restrict__ out) {
  int g = blockIdx.x, o = threadIdx.x;
  float s = bl[o];
  const float* pg = pooled + (size_t)g * 512;
#pragma unroll 4
  for (int k = 0; k < 512; ++k) s += pg[k] * Wl[(size_t)k * 256 + o];
  out[(size_t)g * 256 + o] = s;
}

extern "C" void kernel_launch(void* const* d_in, const int* in_sizes, int n_in,
                              void* d_out, int out_size, void* d_ws, size_t ws_size,
                              hipStream_t stream) {
  const float* x  = (const float*)d_in[0];
  const int*   ei = (const int*)d_in[1];
  const float* ea = (const float*)d_in[2];
  const int* gids = (const int*)d_in[3];
  const float* Wn = (const float*)d_in[4];
  const float* bn = (const float*)d_in[5];
  const float* We = (const float*)d_in[6];
  const float* be = (const float*)d_in[7];
  const float* W1 = (const float*)d_in[8];
  const float* b1 = (const float*)d_in[9];
  const float* W2 = (const float*)d_in[10];
  const float* b2 = (const float*)d_in[11];
  const float* Wl = (const float*)d_in[12];
  const float* bl = (const float*)d_in[13];
  float* out = (float*)d_out;

  char* ws = (char*)d_ws;
  size_t off = 0;
  auto take = [&](size_t bytes) -> void* {
    void* p = ws + off;
    off += (bytes + 255) & ~(size_t)255;
    return p;
  };
  bf16*  Wp    = (bf16*)take(2 * 32768);                   // W1p || W2p (64 KB)
  bf16*  Wnp   = (bf16*)take(8192);
  bf16*  Wep   = (bf16*)take(8192);
  bf16*  e     = (bf16*)take((size_t)N_EDGES * 128 * 2);   // bf16 edge stream
  float* h     = (float*)take((size_t)N_NODES * 128 * 4);
  float* agg   = (float*)take((size_t)N_NODES * 128 * 4);
  float* pooled= (float*)take((size_t)N_GRAPHS * 512 * 4);

  pack_w_kernel<<<32, 256, 0, stream>>>(W1, (u32*)Wp,        128, 4, 8192);
  pack_w_kernel<<<32, 256, 0, stream>>>(W2, (u32*)Wp + 8192, 128, 4, 8192);
  pack_w_kernel<<< 8, 256, 0, stream>>>(Wn, (u32*)Wnp,        32, 1, 2048);
  pack_w_kernel<<< 8, 256, 0, stream>>>(We, (u32*)Wep,        16, 1, 2048);
  zero_kernel<<<20000, 256, 0, stream>>>(agg, (long)N_NODES * 128);
  zero_kernel<<<  128, 256, 0, stream>>>(pooled, (long)N_GRAPHS * 512);

  node_transform_kernel<<< 313, 256, 0, stream>>>(x, Wnp, bn, h, 2500);
  edge_transform_kernel<<<5000, 256, 0, stream>>>(ea, Wep, be, e, 40000);

  for (int t = 0; t < T_STEPS; ++t) {
    edge_msg_kernel<<<80000, 256, 0, stream>>>(e, h, ei, agg, N_EDGES);
    mlp_kernel<<<313, 256, 0, stream>>>(Wp, b1, b2, h, agg, gids, pooled, t, 2500);
  }
  final_kernel<<<64, 256, 0, stream>>>(pooled, Wl, bl, out);
}
